// MockAttentionHead_71614284694212
// MI455X (gfx1250) — compile-verified, matrix operand order checked by reading, not run on hardware
//
#include <hip/hip_runtime.h>
#include <hip/hip_bf16.h>
#include <math.h>

typedef __attribute__((ext_vector_type(16))) _Float16 v16h;
typedef __attribute__((ext_vector_type(8)))  _Float16 v8h;
typedef __attribute__((ext_vector_type(8)))  float    v8f;
typedef int v4i_g __attribute__((vector_size(16)));   // matches builtin's param type

#define DEPS 1e-8f
constexpr int BN = 4096;   // batch
constexpr int DD = 128;    // feature dim
constexpr float TEMP = 11.313708498984760390f; // sqrt(128)

#define WMMA_F16(a, b, c) \
  __builtin_amdgcn_wmma_f32_16x16x32_f16(false, (a), false, (b), (short)0, (c), false, false)

// ---------------------------------------------------------------------------
// Async global->LDS copy (CDNA5 GLOBAL_LOAD_ASYNC_TO_LDS_B128, ASYNCcnt),
// with a plain vector-copy fallback if the builtin is absent.
// Builtin signature (from hipcc diagnostic): param0 = v4i addrspace(1)*,
// param1 = LDS side, then imm offset + imm cpol.
// ---------------------------------------------------------------------------
#if defined(__HIP_DEVICE_COMPILE__) && __has_builtin(__builtin_amdgcn_global_load_async_to_lds_b128)
#define HAVE_ASYNC_COPY 1
#else
#define HAVE_ASYNC_COPY 0
#endif

__device__ __forceinline__ void copy16_g2l(const _Float16* __restrict__ g,
                                           _Float16* __restrict__ l) {
#if HAVE_ASYNC_COPY
  __builtin_amdgcn_global_load_async_to_lds_b128(
      (__attribute__((address_space(1))) v4i_g*)g,
      (__attribute__((address_space(3))) v4i_g*)l, 0, 0);
#else
  *(v8h*)l = *(const v8h*)g;
#endif
}

__device__ __forceinline__ void async_wait_all() {
#if HAVE_ASYNC_COPY
#if __has_builtin(__builtin_amdgcn_s_wait_asynccnt)
  __builtin_amdgcn_s_wait_asynccnt(0);
#else
  asm volatile("s_wait_asynccnt 0x0" ::: "memory");
#endif
#endif
}

// ---------------------------------------------------------------------------
// WMMA tile loaders. CDNA5 16-bit A-matrix 16x32 layout (CDNA5 ISA):
//   lane t (0..31): row m = t&15; half-select h = t>>4
//   VGPR0..3 hold K = 8*h + {0..7};  VGPR4..7 hold K = 16 + 8*h + {0..7}
// i.e. two contiguous 8-element (16-byte) chunks per lane.
// B (KxN) mirrors this with lane = column n, so for C = A * B with
// B[k][n] = R[n][k] (Gram / W^T patterns) the SAME loader works on R.
// ---------------------------------------------------------------------------
__device__ __forceinline__ v16h load_tile_h(const _Float16* __restrict__ base,
                                            int ld, int lane) {
  const int m  = lane & 15;
  const int kb = (lane >> 4) << 3;               // 0 or 8
  const _Float16* r = base + (size_t)m * ld + kb;
  v8h lo = *(const v8h*)(r);                     // K = kb .. kb+7
  v8h hi = *(const v8h*)(r + 16);                // K = 16+kb .. 16+kb+7
  return __builtin_shufflevector(lo, hi, 0,1,2,3,4,5,6,7,8,9,10,11,12,13,14,15);
}

__device__ __forceinline__ v16h load_tile_f32(const float* __restrict__ base,
                                              int ld, int lane) {
  const int m  = lane & 15;
  const int kb = (lane >> 4) << 3;
  const float* r = base + (size_t)m * ld + kb;
  v16h o;
#pragma unroll
  for (int e = 0; e < 8; ++e) o[e]     = (_Float16)r[e];
#pragma unroll
  for (int e = 0; e < 8; ++e) o[8 + e] = (_Float16)r[16 + e];
  return o;
}

__device__ __forceinline__ float wave_reduce_sum(float v) {
#pragma unroll
  for (int off = 16; off > 0; off >>= 1) v += __shfl_down(v, off, 32);
  return v;
}

// ---------------------------------------------------------------------------
// 1) Projection: P[B,D] = X[B,D] * W[D,D]^T + bias.   One wave per 16x16 tile.
// ---------------------------------------------------------------------------
__global__ void proj_kernel(const float* __restrict__ X, const float* __restrict__ W,
                            const float* __restrict__ bias, float* __restrict__ P) {
  const int lane = threadIdx.x;
  const int i0 = blockIdx.x * 16;
  const int j0 = blockIdx.y * 16;
  const int n  = lane & 15;
  const float bv = bias[j0 + n];
  v8f c;
#pragma unroll
  for (int r = 0; r < 8; ++r) c[r] = bv;
#pragma unroll
  for (int kk = 0; kk < DD; kk += 32) {
    v16h a = load_tile_f32(X + (size_t)i0 * DD + kk, DD, lane);
    v16h b = load_tile_f32(W + (size_t)j0 * DD + kk, DD, lane);
    c = WMMA_F16(a, b, c);
  }
  const int mbase = i0 + ((lane & 16) ? 8 : 0);
#pragma unroll
  for (int r = 0; r < 8; ++r)
    P[(size_t)(mbase + r) * DD + j0 + n] = c[r];
}

// ---------------------------------------------------------------------------
// 2) Per-row scalars + rescale to f16.
//    metric_mode=1: alpha from the collapsed metric normalization.
//    metric_mode=0: alpha=1 (raw input copies for the input-dist Gram).
// ---------------------------------------------------------------------------
__global__ void rowscale_kernel(const float* __restrict__ P,
                                _Float16* __restrict__ O16,
                                float* __restrict__ rsq, int metric_mode) {
  __shared__ float red[DD];
  const int row = blockIdx.x;
  const int t   = threadIdx.x;
  const float v = P[(size_t)row * DD + t];
  red[t] = v * v;
  __syncthreads();
#pragma unroll
  for (int s = DD / 2; s > 0; s >>= 1) {
    if (t < s) red[t] += red[t + s];
    __syncthreads();
  }
  const float s = red[0];
  float alpha = 1.0f;
  if (metric_mode) {
    const float fd  = (float)DD;
    const float fro = sqrtf(s * s / (fd * fd) + 2.0f * s / fd + fd);
    const float qn2 = (s * s / fd + s) / (fro + DEPS);
    const float qn  = sqrtf(fmaxf(qn2, 0.0f));
    alpha = 1.0f / (qn + DEPS);
  }
  O16[(size_t)row * DD + t] = (_Float16)(v * alpha);
  if (t == 0) rsq[row] = s * alpha * alpha;
}

// ---------------------------------------------------------------------------
// 3) dists -> sims*T, written to S[B,B].  One wave per 16x16 tile, K=128.
// ---------------------------------------------------------------------------
__global__ void dist_sim_kernel(const _Float16* __restrict__ A16,
                                const _Float16* __restrict__ B16,
                                const float* __restrict__ asq,
                                const float* __restrict__ bsq,
                                float* __restrict__ S) {
  const int lane = threadIdx.x;
  const int i0 = blockIdx.y * 16;
  const int j0 = blockIdx.x * 16;
  v8f c = {};
#pragma unroll
  for (int kk = 0; kk < DD; kk += 32) {
    v16h a = load_tile_h(A16 + (size_t)i0 * DD + kk, DD, lane);
    v16h b = load_tile_h(B16 + (size_t)j0 * DD + kk, DD, lane);
    c = WMMA_F16(a, b, c);
  }
  const int n = lane & 15;
  const int mbase = i0 + ((lane & 16) ? 8 : 0);
  const float bn = bsq[j0 + n];
#pragma unroll
  for (int r = 0; r < 8; ++r) {
    const float d2 = asq[mbase + r] + bn - 2.0f * c[r];
    const float d  = sqrtf(fmaxf(d2, 0.0f));
    S[(size_t)(mbase + r) * BN + j0 + n] = TEMP / (1.0f + d);
  }
}

// ---------------------------------------------------------------------------
// 4) Row softmax in-place + f16 copy + ||p||^2 per row.  256 thr, 16 elts/thr.
// ---------------------------------------------------------------------------
__global__ void softmax_kernel(float* __restrict__ S, _Float16* __restrict__ P16,
                               float* __restrict__ psq) {
  __shared__ float red[256];
  const int row = blockIdx.x;
  const int t   = threadIdx.x;
  float* rowp = S + (size_t)row * BN;
  float vals[16];
  float mx = -1e30f;
#pragma unroll
  for (int e = 0; e < 16; ++e) { vals[e] = rowp[t + e * 256]; mx = fmaxf(mx, vals[e]); }
  red[t] = mx; __syncthreads();
#pragma unroll
  for (int s = 128; s > 0; s >>= 1) { if (t < s) red[t] = fmaxf(red[t], red[t + s]); __syncthreads(); }
  mx = red[0];
  __syncthreads();
  float sum = 0.0f;
#pragma unroll
  for (int e = 0; e < 16; ++e) { vals[e] = __expf(vals[e] - mx); sum += vals[e]; }
  red[t] = sum; __syncthreads();
#pragma unroll
  for (int s = 128; s > 0; s >>= 1) { if (t < s) red[t] += red[t + s]; __syncthreads(); }
  const float inv = 1.0f / red[0];
  __syncthreads();
  float sq = 0.0f;
#pragma unroll
  for (int e = 0; e < 16; ++e) {
    const float p = vals[e] * inv;
    rowp[t + e * 256] = p;
    P16[(size_t)row * BN + t + e * 256] = (_Float16)p;
    sq += p * p;
  }
  red[t] = sq; __syncthreads();
#pragma unroll
  for (int s = 128; s > 0; s >>= 1) { if (t < s) red[t] += red[t + s]; __syncthreads(); }
  if (t == 0) psq[row] = red[0];
}

// ---------------------------------------------------------------------------
// 5a) mean(cdist) for K=128 (input dists): one wave per 16x16 tile.
// ---------------------------------------------------------------------------
__global__ void gram_mean_small_kernel(const _Float16* __restrict__ A16,
                                       const _Float16* __restrict__ B16,
                                       const float* __restrict__ asq,
                                       const float* __restrict__ bsq,
                                       float* __restrict__ accum) {
  const int lane = threadIdx.x;
  const int i0 = blockIdx.y * 16;
  const int j0 = blockIdx.x * 16;
  v8f c = {};
#pragma unroll
  for (int kk = 0; kk < DD; kk += 32) {
    v16h a = load_tile_h(A16 + (size_t)i0 * DD + kk, DD, lane);
    v16h b = load_tile_h(B16 + (size_t)j0 * DD + kk, DD, lane);
    c = WMMA_F16(a, b, c);
  }
  const int n = lane & 15;
  const int mbase = i0 + ((lane & 16) ? 8 : 0);
  const float bn = bsq[j0 + n];
  float s = 0.0f;
#pragma unroll
  for (int r = 0; r < 8; ++r) {
    const float d2 = asq[mbase + r] + bn - 2.0f * c[r];
    s += sqrtf(fmaxf(d2, 0.0f));
  }
  s = wave_reduce_sum(s);
  if (lane == 0) atomicAdd(accum, s);
}

// ---------------------------------------------------------------------------
// 5b) mean(score_dists): the dominant 4096^3 Gram.
//     Workgroup (256 thr, 8 waves) computes a 128x64 block of P*P^T.
//     Per K-step: async-stage A(128x32) + B(64x32) slices into LDS, then each
//     wave does a 32x32 register tile = 2x2 WMMA accumulators.
// ---------------------------------------------------------------------------
__global__ void __launch_bounds__(256)
score_gram_kernel(const _Float16* __restrict__ P16, const float* __restrict__ psq,
                  float* __restrict__ accum) {
  __shared__ __align__(16) _Float16 lA[128 * 32];   // 8 KB
  __shared__ __align__(16) _Float16 lB[64 * 32];    // 4 KB
  const int t    = threadIdx.x;
  const int lane = t & 31;
  const int w    = t >> 5;     // wave 0..7
  const int wi   = w & 3;      // 32-row sub-block
  const int wj   = w >> 2;     // 32-col sub-block
  const int i0 = blockIdx.y * 128;
  const int j0 = blockIdx.x * 64;

  v8f c00 = {}, c01 = {}, c10 = {}, c11 = {};

  for (int kk = 0; kk < BN; kk += 32) {
    // stage A block: 128 rows x 32 halfs = 512 x 16B chunks (2 per thread)
#pragma unroll
    for (int it = 0; it < 2; ++it) {
      const int chunk = it * 256 + t;
      const int r = chunk >> 2, cx = chunk & 3;
      copy16_g2l(P16 + (size_t)(i0 + r) * BN + kk + cx * 8, &lA[r * 32 + cx * 8]);
    }
    // stage B block: 64 rows x 32 halfs = 256 x 16B chunks (1 per thread)
    {
      const int r = t >> 2, cx = t & 3;
      copy16_g2l(P16 + (size_t)(j0 + r) * BN + kk + cx * 8, &lB[r * 32 + cx * 8]);
    }
    async_wait_all();
    __syncthreads();

    v16h a0 = load_tile_h(&lA[(wi * 32 + 0)  * 32], 32, lane);
    v16h a1 = load_tile_h(&lA[(wi * 32 + 16) * 32], 32, lane);
    v16h b0 = load_tile_h(&lB[(wj * 32 + 0)  * 32], 32, lane);
    v16h b1 = load_tile_h(&lB[(wj * 32 + 16) * 32], 32, lane);
    c00 = WMMA_F16(a0, b0, c00);
    c01 = WMMA_F16(a0, b1, c01);
    c10 = WMMA_F16(a1, b0, c10);
    c11 = WMMA_F16(a1, b1, c11);
    __syncthreads();
  }

  // epilogue: sum of sqrt(psq_i + psq_j - 2*g) over the 32x32 tile
  const int n  = lane & 15;
  const int mo = (lane & 16) ? 8 : 0;
  const int rbase = i0 + wi * 32;
  const int cbase = j0 + wj * 32;
  const float bn0 = psq[cbase + n];
  const float bn1 = psq[cbase + 16 + n];
  float s = 0.0f;
#pragma unroll
  for (int r = 0; r < 8; ++r) {
    const float a0v = psq[rbase + mo + r];
    const float a1v = psq[rbase + 16 + mo + r];
    s += sqrtf(fmaxf(a0v + bn0 - 2.0f * c00[r], 0.0f));
    s += sqrtf(fmaxf(a0v + bn1 - 2.0f * c01[r], 0.0f));
    s += sqrtf(fmaxf(a1v + bn0 - 2.0f * c10[r], 0.0f));
    s += sqrtf(fmaxf(a1v + bn1 - 2.0f * c11[r], 0.0f));
  }
  s = wave_reduce_sum(s);
  if (lane == 0) atomicAdd(accum, s);
}

// ---------------------------------------------------------------------------
// 6) Finalize: scale = meanIn/(meanScore+eps); out *= scale/(scale+eps)
//    (softmax rows sum to 1, so the renormalization reduces to this factor).
// ---------------------------------------------------------------------------
__global__ void finalize_kernel(float* __restrict__ out,
                                const float* __restrict__ sums) {
  const float n2     = (float)((size_t)BN * BN);
  const float meanIn = sums[0] / n2;
  const float meanSc = sums[1] / n2;
  const float scale  = meanIn / (meanSc + DEPS);
  const float f      = scale / (scale + DEPS);
  const size_t idx = (size_t)blockIdx.x * blockDim.x + threadIdx.x;
  out[idx] *= f;
}

__global__ void zero2_kernel(float* __restrict__ p) {
  if (threadIdx.x < 2) p[threadIdx.x] = 0.0f;
}

// ---------------------------------------------------------------------------
extern "C" void kernel_launch(void* const* d_in, const int* in_sizes, int n_in,
                              void* d_out, int out_size, void* d_ws, size_t ws_size,
                              hipStream_t stream) {
  const float* Xq = (const float*)d_in[0];
  const float* Xk = (const float*)d_in[1];
  const float* Wq = (const float*)d_in[2];
  const float* bq = (const float*)d_in[3];
  const float* Wk = (const float*)d_in[4];
  const float* bk = (const float*)d_in[5];
  float* out = (float*)d_out;

  char* ws = (char*)d_ws;
  const size_t MB = 1u << 20, KB = 1u << 10;
  float*     qf   = (float*)    (ws + 0 * MB);            // 2 MB
  float*     kf   = (float*)    (ws + 2 * MB);            // 2 MB
  _Float16*  qn16 = (_Float16*) (ws + 4 * MB);            // 1 MB
  _Float16*  kn16 = (_Float16*) (ws + 5 * MB);            // 1 MB
  _Float16*  xq16 = (_Float16*) (ws + 6 * MB);            // 1 MB
  _Float16*  xk16 = (_Float16*) (ws + 7 * MB);            // 1 MB
  float*     qsq  = (float*)    (ws + 8 * MB);            // 16 KB
  float*     ksq  = (float*)    (ws + 8 * MB + 16 * KB);
  float*     xqsq = (float*)    (ws + 8 * MB + 32 * KB);
  float*     xksq = (float*)    (ws + 8 * MB + 48 * KB);
  float*     sums = (float*)    (ws + 8 * MB + 64 * KB);  // 2 floats
  float*     psq  = (float*)    (ws + 8 * MB + 80 * KB);  // 16 KB
  _Float16*  p16  = (_Float16*) (ws + 9 * MB);            // 32 MB

  zero2_kernel<<<1, 32, 0, stream>>>(sums);

  // projections (WMMA f16, f32 accum)
  proj_kernel<<<dim3(BN / 16, DD / 16), 32, 0, stream>>>(Xq, Wq, bq, qf);
  proj_kernel<<<dim3(BN / 16, DD / 16), 32, 0, stream>>>(Xk, Wk, bk, kf);

  // per-row metric normalization (collapsed to scalars) + f16 copies
  rowscale_kernel<<<BN, DD, 0, stream>>>(qf, qn16, qsq, 1);
  rowscale_kernel<<<BN, DD, 0, stream>>>(kf, kn16, ksq, 1);
  rowscale_kernel<<<BN, DD, 0, stream>>>(Xq, xq16, xqsq, 0);
  rowscale_kernel<<<BN, DD, 0, stream>>>(Xk, xk16, xksq, 0);

  // sims * temperature into d_out
  dist_sim_kernel<<<dim3(BN / 16, BN / 16), 32, 0, stream>>>(qn16, kn16, qsq, ksq, out);

  // row softmax (in place) + f16 scores + row norms
  softmax_kernel<<<BN, 256, 0, stream>>>(out, p16, psq);

  // mean(input_dists) -> sums[0]   (K = 128)
  gram_mean_small_kernel<<<dim3(BN / 16, BN / 16), 32, 0, stream>>>(
      xq16, xk16, xqsq, xksq, &sums[0]);
  // mean(score_dists) -> sums[1]   (K = 4096, async-LDS-staged blocked WMMA)
  score_gram_kernel<<<dim3(BN / 64, BN / 128), 256, 0, stream>>>(p16, psq, &sums[1]);

  // out *= scale/(scale+eps)
  finalize_kernel<<<(unsigned)(((size_t)BN * BN) / 256), 256, 0, stream>>>(out, sums);
}
